// ReactionGeneProcessor_39779987096426
// MI455X (gfx1250) — matrix operand order repaired and checked by reading, not compiled, not run on hardware
//
#include <hip/hip_runtime.h>
#include <hip/hip_bf16.h>

typedef _Float16 half_t;
typedef __attribute__((ext_vector_type(16))) _Float16 v16h;
typedef __attribute__((ext_vector_type(8)))  _Float16 v8h;
typedef __attribute__((ext_vector_type(4)))  _Float16 v4h;
typedef __attribute__((ext_vector_type(8)))  float    v8f;

#define C_DIM 512
#define L_DIM 32
#define HEADS 4
#define DH 128
#define CC (C_DIM * C_DIM)
#define RSQRT_DH 0.088388347648318447f

// ---------------- LDS layout (bytes) ----------------
// x32  : float[32][512]   0      .. 65536
// a16  : half [32][512]   65536  .. 98304
// q16  : half [32][512]   98304  .. 131072
// k16  : half [32][512]   131072 .. 163840
// vT16 : half [512][32]   163840 .. 196608
// t32  : float[32][512]   196608 .. 262144
// sc   : float[4][32][32] 262144 .. 278528
// p16  : half [4][32][32] 278528 .. 286720
#define SMEM_BYTES 286720

// ---------------------------------------------------------------------------
// WMMA fragment loaders (CDNA5 16x16x32 f16, wave32). 16-byte vector loads.
// A (16x32, MxK): lane l -> row m0+(l&15); halves 0..7 cover K = k0+grp*8 .. +7,
//                 halves 8..15 cover K = k0+16+grp*8 .. +7   (grp = l>>4)
// B (32x16, KxN): lane l -> col n0+(l&15); halves 0..7 cover K = k0+grp*8 .. +7,
//                 halves 8..15 cover K = k0+16+grp*8 .. +7
// ---------------------------------------------------------------------------
__device__ __forceinline__ v16h load_frag_a(const half_t* __restrict__ src, int ld,
                                            int m0, int k0, int lane) {
  int grp = lane >> 4;
  const half_t* p = src + (m0 + (lane & 15)) * ld + k0 + grp * 8;
  v8h lo = *(const v8h*)p;
  v8h hi = *(const v8h*)(p + 16);
  v16h r;
#pragma unroll
  for (int j = 0; j < 8; ++j) { r[j] = lo[j]; r[j + 8] = hi[j]; }
  return r;
}

__device__ __forceinline__ v16h load_frag_b(const half_t* __restrict__ src, int ld,
                                            int n0, int k0, int lane) {
  int grp = lane >> 4;
  const half_t* p = src + (n0 + (lane & 15)) * ld + k0 + grp * 8;
  v8h lo = *(const v8h*)p;
  v8h hi = *(const v8h*)(p + 16);
  v16h r;
#pragma unroll
  for (int j = 0; j < 8; ++j) { r[j] = lo[j]; r[j + 8] = hi[j]; }
  return r;
}

__device__ __forceinline__ v8f wmma_f16(v16h a, v16h b, v8f c) {
  return __builtin_amdgcn_wmma_f32_16x16x32_f16(false, a, false, b, (short)0, c,
                                                false, false);
}

// ---------------------------------------------------------------------------
// [32,512] = A[32,512] @ W^T (W row-major [n][k], f16) + bias, tiled WMMA.
// Each wave owns BOTH 16-row M-tiles for its 4 N-tiles: every B (weight)
// fragment is loaded from L2 exactly once per workgroup and feeds 2 WMMAs,
// halving weight traffic vs. the mt-split mapping. A fragments for both
// M-tiles are cached in VGPRs (32 x v16h = 256 VGPRs; wave32 allows 1024,
// and LDS caps occupancy at 2 waves/SIMD anyway).
// ---------------------------------------------------------------------------
__device__ __forceinline__ void gemm_proj(const half_t* __restrict__ A,
                                          const half_t* __restrict__ W,
                                          const float* __restrict__ bias,
                                          float* outF, half_t* outH, half_t* outT,
                                          bool relu, int tid) {
  int wave = tid >> 5, lane = tid & 31, grp = lane >> 4;
  v16h af0[16], af1[16];
#pragma unroll
  for (int kc = 0; kc < 16; ++kc) {
    af0[kc] = load_frag_a(A, C_DIM, 0,  kc * 32, lane);
    af1[kc] = load_frag_a(A, C_DIM, 16, kc * 32, lane);
  }
  for (int nt = wave; nt < 32; nt += 8) {
    int n0 = nt * 16;
    v8f acc0 = {}, acc1 = {};
#pragma unroll
    for (int kc = 0; kc < 16; ++kc) {
      v16h bf = load_frag_b(W, C_DIM, n0, kc * 32, lane);
      acc0 = wmma_f16(af0[kc], bf, acc0);
      acc1 = wmma_f16(af1[kc], bf, acc1);
    }
    int n = n0 + (lane & 15);
    float bv = bias ? bias[n] : 0.f;
#pragma unroll
    for (int j = 0; j < 8; ++j) {
      int m = grp * 8 + j;
      float v0 = acc0[j] + bv;
      float v1 = acc1[j] + bv;
      if (relu) { v0 = fmaxf(v0, 0.f); v1 = fmaxf(v1, 0.f); }
      if (outF) { outF[m * C_DIM + n] = v0; outF[(m + 16) * C_DIM + n] = v1; }
      if (outH) { outH[m * C_DIM + n] = (half_t)v0; outH[(m + 16) * C_DIM + n] = (half_t)v1; }
      if (outT) { outT[n * L_DIM + m] = (half_t)v0; outT[n * L_DIM + m + 16] = (half_t)v1; }
    }
  }
}

// LayerNorm over 32 rows of 512: x = LN(x + (rowvalid ? t : 0)) * g + b; also f16 copy.
__device__ __forceinline__ void ln_rows32(const float* __restrict__ t,
                                          float* __restrict__ x, half_t* __restrict__ h16,
                                          const float* __restrict__ g,
                                          const float* __restrict__ b,
                                          int len, bool usemask, int tid) {
  int wave = tid >> 5, lane = tid & 31;
  for (int row = wave; row < L_DIM; row += 8) {
    bool rv = (!usemask) || (row < len);
    float vals[16];
    float s = 0.f, s2 = 0.f;
#pragma unroll
    for (int k = 0; k < 16; ++k) {
      int c = lane + k * 32;
      float v = x[row * C_DIM + c] + (rv ? t[row * C_DIM + c] : 0.f);
      vals[k] = v; s += v; s2 += v * v;
    }
#pragma unroll
    for (int off = 16; off > 0; off >>= 1) {
      s  += __shfl_xor(s, off, 32);
      s2 += __shfl_xor(s2, off, 32);
    }
    float mu = s * (1.f / 512.f);
    float var = s2 * (1.f / 512.f) - mu * mu;
    float inv = rsqrtf(var + 1e-5f);
#pragma unroll
    for (int k = 0; k < 16; ++k) {
      int c = lane + k * 32;
      float o = (vals[k] - mu) * inv * g[c] + b[c];
      x[row * C_DIM + c] = o;
      h16[row * C_DIM + c] = (half_t)o;
    }
  }
}

// Block-wide LayerNorm over one row of 512 (in != out).
__device__ __forceinline__ void ln_row1(const float* __restrict__ in,
                                        float* __restrict__ out,
                                        const float* __restrict__ g,
                                        const float* __restrict__ b,
                                        float* __restrict__ red, int tid) {
  float s = 0.f, s2 = 0.f;
  for (int c = tid; c < C_DIM; c += 256) { float v = in[c]; s += v; s2 += v * v; }
#pragma unroll
  for (int off = 16; off > 0; off >>= 1) {
    s  += __shfl_xor(s, off, 32);
    s2 += __shfl_xor(s2, off, 32);
  }
  int wave = tid >> 5, lane = tid & 31;
  if (lane == 0) { red[wave] = s; red[8 + wave] = s2; }
  __syncthreads();
  if (tid == 0) {
    float a = 0.f, bb = 0.f;
    for (int w = 0; w < 8; ++w) { a += red[w]; bb += red[8 + w]; }
    red[16] = a; red[17] = bb;
  }
  __syncthreads();
  float mu = red[16] * (1.f / 512.f);
  float var = red[17] * (1.f / 512.f) - mu * mu;
  float inv = rsqrtf(var + 1e-5f);
  for (int c = tid; c < C_DIM; c += 256) out[c] = (in[c] - mu) * inv * g[c] + b[c];
  __syncthreads();
}

// f32 accumulate dot of 512-wide f16 weight row with f32 LDS vector.
__device__ __forceinline__ float dot512_h(const half_t* __restrict__ w,
                                          const float* __restrict__ xl) {
  float s = 0.f;
  for (int k = 0; k < C_DIM; k += 8) {
    v8h wv = *(const v8h*)(w + k);
#pragma unroll
    for (int j = 0; j < 8; ++j) s += (float)wv[j] * xl[k + j];
  }
  return s;
}

__global__ void cvt_f32_f16_kernel(const float* __restrict__ src,
                                   half_t* __restrict__ dst, int n) {
  int i = blockIdx.x * blockDim.x + threadIdx.x;
  int stride = gridDim.x * blockDim.x;
  for (; i < n; i += stride) dst[i] = (half_t)src[i];
}

// ---------------------------------------------------------------------------
// Main kernel: one workgroup (256 threads = 8 wave32) per reaction.
// ---------------------------------------------------------------------------
__global__ __launch_bounds__(256)
void settransformer_kernel(const float* __restrict__ gene,
                           const int* __restrict__ gidx,
                           const int* __restrict__ glen,
                           const half_t* __restrict__ Wq16,
                           const half_t* __restrict__ Wk16,
                           const half_t* __restrict__ Wv16,
                           const half_t* __restrict__ Wo16,
                           const half_t* __restrict__ ff16,
                           const half_t* __restrict__ pma16,
                           const float* __restrict__ bq, const float* __restrict__ bk,
                           const float* __restrict__ bv, const float* __restrict__ bo,
                           const float* __restrict__ ffb,
                           const float* __restrict__ g1, const float* __restrict__ b1,
                           const float* __restrict__ g2, const float* __restrict__ b2,
                           const float* __restrict__ pma_b,
                           const float* __restrict__ seed,
                           float* __restrict__ out) {
  extern __shared__ char smem[];
  float*  x32 = (float*)(smem);
  half_t* a16 = (half_t*)(smem + 65536);
  half_t* q16 = (half_t*)(smem + 98304);
  half_t* k16 = (half_t*)(smem + 131072);
  half_t* vT  = (half_t*)(smem + 163840);
  float*  t32 = (float*)(smem + 196608);
  float*  sc  = (float*)(smem + 262144);
  half_t* p16 = (half_t*)(smem + 278528);

  const int r = blockIdx.x;
  const int tid = threadIdx.x;
  const int wave = tid >> 5, lane = tid & 31, grp = lane >> 4;
  const int len = glen[r];

  // ---- Gather: x = where(mask, gene[idx], 0), f32 + f16 copies ----
  {
    int row = tid >> 3;
    int c0 = (tid & 7) * 64;
    int gi = gidx[(size_t)r * L_DIM + row];
    bool rv = row < len;
    const float* src = gene + (size_t)gi * C_DIM;
    for (int c = c0; c < c0 + 64; c += 4) {
      float4 vv = rv ? *(const float4*)(src + c) : make_float4(0.f, 0.f, 0.f, 0.f);
      *(float4*)(x32 + row * C_DIM + c) = vv;
      v4h hv = {(half_t)vv.x, (half_t)vv.y, (half_t)vv.z, (half_t)vv.w};
      *(v4h*)(a16 + row * C_DIM + c) = hv;
    }
  }
  __syncthreads();

  // ---- Encoder SAB blocks 0 and 1 ----
  for (int blk = 0; blk < 2; ++blk) {
    const size_t wo = (size_t)blk * CC;
    const int bo_ = blk * C_DIM;
    gemm_proj(a16, Wq16 + wo, bq + bo_, nullptr, q16, nullptr, false, tid);
    gemm_proj(a16, Wk16 + wo, bk + bo_, nullptr, k16, nullptr, false, tid);
    gemm_proj(a16, Wv16 + wo, bv + bo_, nullptr, nullptr, vT, false, tid);
    __syncthreads();

    // scores[h][q][k] = (Q_h @ K_h^T)/sqrt(Dh), key-masked  (16 WMMA tiles)
    for (int rep = 0; rep < 2; ++rep) {
      int t = wave * 2 + rep;
      int h = t >> 2, mt = (t >> 1) & 1, nt = t & 1;
      v8f acc = {};
#pragma unroll
      for (int kc = 0; kc < 4; ++kc) {
        v16h af = load_frag_a(q16 + h * DH, C_DIM, mt * 16, kc * 32, lane);
        v16h bf = load_frag_b(k16 + h * DH, C_DIM, nt * 16, kc * 32, lane);
        acc = wmma_f16(af, bf, acc);
      }
      int n = nt * 16 + (lane & 15);
#pragma unroll
      for (int j = 0; j < 8; ++j) {
        int m = mt * 16 + grp * 8 + j;
        float v = acc[j] * RSQRT_DH;
        if (n >= len) v = -1e30f;
        sc[(h * 32 + m) * 32 + n] = v;
      }
    }
    __syncthreads();

    // softmax over keys -> p16
    if (tid < 128) {
      int h = tid >> 5, qr = tid & 31;
      float* rowp = sc + (h * 32 + qr) * 32;
      float mx = -1e30f;
      for (int k = 0; k < 32; ++k) mx = fmaxf(mx, rowp[k]);
      float e[32], ss = 0.f;
      for (int k = 0; k < 32; ++k) { e[k] = __expf(rowp[k] - mx); ss += e[k]; }
      float inv = 1.f / ss;
      for (int k = 0; k < 32; ++k) p16[(h * 32 + qr) * 32 + k] = (half_t)(e[k] * inv);
    }
    __syncthreads();

    // attn @ V -> a16 (f16): both M-tiles per wave, B (vT) loaded once.
    {
      for (int nt = wave; nt < 32; nt += 8) {
        int n0 = nt * 16;
        int h = nt >> 3;
        v16h bf  = load_frag_b(vT, 32, n0, 0, lane);
        v16h af0 = load_frag_a(p16 + h * 32 * 32, 32, 0,  0, lane);
        v16h af1 = load_frag_a(p16 + h * 32 * 32, 32, 16, 0, lane);
        v8f acc0 = {}, acc1 = {};
        acc0 = wmma_f16(af0, bf, acc0);
        acc1 = wmma_f16(af1, bf, acc1);
        int n = n0 + (lane & 15);
#pragma unroll
        for (int j = 0; j < 8; ++j) {
          int m = grp * 8 + j;
          a16[m * C_DIM + n] = (half_t)acc0[j];
          a16[(m + 16) * C_DIM + n] = (half_t)acc1[j];
        }
      }
    }
    __syncthreads();

    // O-proj, mask padded query rows, residual + LN1, FF + ReLU, residual + LN2
    gemm_proj(a16, Wo16 + wo, bo + bo_, t32, nullptr, nullptr, false, tid);
    __syncthreads();
    ln_rows32(t32, x32, a16, g1 + bo_, b1 + bo_, len, true, tid);
    __syncthreads();
    gemm_proj(a16, ff16 + wo, ffb + bo_, t32, nullptr, nullptr, true, tid);
    __syncthreads();
    ln_rows32(t32, x32, a16, g2 + bo_, b2 + bo_, len, false, tid);
    __syncthreads();
  }

  // ---- PMA: y = relu(x @ pmaW^T + b); seed attends over y ----
  gemm_proj(a16, pma16, pma_b, nullptr, q16, nullptr, true, tid);  // y -> q16
  __syncthreads();
  gemm_proj(q16, Wk16 + 2 * (size_t)CC, bk + 2 * C_DIM, nullptr, k16, nullptr, false, tid);
  gemm_proj(q16, Wv16 + 2 * (size_t)CC, bv + 2 * C_DIM, nullptr, nullptr, vT, false, tid);

  float* F = t32;                 // scalar-stage float scratch
  float* seedL = F;               // 512
  float* qs    = F + 512;         // 512
  float* scp   = F + 1024;        // 128
  float* pp    = F + 1152;        // 128
  float* ao    = F + 1280;        // 512
  float* xs    = F + 1792;        // 512
  float* tva   = F + 2304;        // 512
  float* tvb   = F + 2880;        // 512
  float* red   = F + 3456;        // 32

  for (int c = tid; c < C_DIM; c += 256) seedL[c] = seed[c];
  __syncthreads();

  // seed query projection (1x512 @ 512x512)
  for (int c = tid; c < C_DIM; c += 256)
    qs[c] = bq[2 * C_DIM + c] + dot512_h(Wq16 + 2 * (size_t)CC + (size_t)c * C_DIM, seedL);
  __syncthreads();

  if (tid < 128) {   // scores [4][32], key-masked
    int h = tid >> 5, kk = tid & 31;
    float s = 0.f;
    for (int d = 0; d < DH; ++d) s += qs[h * DH + d] * (float)k16[kk * C_DIM + h * DH + d];
    s *= RSQRT_DH;
    scp[h * 32 + kk] = (kk < len) ? s : -1e30f;
  }
  __syncthreads();
  if (tid < HEADS) { // softmax per head
    float mx = -1e30f;
    for (int k = 0; k < 32; ++k) mx = fmaxf(mx, scp[tid * 32 + k]);
    float ss = 0.f;
    for (int k = 0; k < 32; ++k) { float e = __expf(scp[tid * 32 + k] - mx); pp[tid * 32 + k] = e; ss += e; }
    float inv = 1.f / ss;
    for (int k = 0; k < 32; ++k) pp[tid * 32 + k] *= inv;
  }
  __syncthreads();
  for (int c = tid; c < C_DIM; c += 256) {  // attn @ V (1 query)
    int h = c >> 7;
    float s = 0.f;
    for (int k = 0; k < 32; ++k) s += pp[h * 32 + k] * (float)vT[c * L_DIM + k];
    ao[c] = s;
  }
  __syncthreads();
  for (int c = tid; c < C_DIM; c += 256)    // O-proj + residual(seed)
    tva[c] = bo[2 * C_DIM + c] + dot512_h(Wo16 + 2 * (size_t)CC + (size_t)c * C_DIM, ao) + seedL[c];
  __syncthreads();
  ln_row1(tva, xs, g1 + 2 * C_DIM, b1 + 2 * C_DIM, red, tid);
  for (int c = tid; c < C_DIM; c += 256)
    tva[c] = xs[c] + fmaxf(0.f, ffb[2 * C_DIM + c] +
                           dot512_h(ff16 + 2 * (size_t)CC + (size_t)c * C_DIM, xs));
  __syncthreads();
  ln_row1(tva, xs, g2 + 2 * C_DIM, b2 + 2 * C_DIM, red, tid);

  // ---- Decoder SAB on single token: softmax over 1 key == 1, attn-out == V ----
  for (int c = tid; c < C_DIM; c += 256)
    tvb[c] = bv[3 * C_DIM + c] + dot512_h(Wv16 + 3 * (size_t)CC + (size_t)c * C_DIM, xs);
  __syncthreads();
  for (int c = tid; c < C_DIM; c += 256)
    tva[c] = xs[c] + bo[3 * C_DIM + c] + dot512_h(Wo16 + 3 * (size_t)CC + (size_t)c * C_DIM, tvb);
  __syncthreads();
  ln_row1(tva, xs, g1 + 3 * C_DIM, b1 + 3 * C_DIM, red, tid);
  for (int c = tid; c < C_DIM; c += 256)
    tva[c] = xs[c] + fmaxf(0.f, ffb[3 * C_DIM + c] +
                           dot512_h(ff16 + 3 * (size_t)CC + (size_t)c * C_DIM, xs));
  __syncthreads();
  ln_row1(tva, xs, g2 + 3 * C_DIM, b2 + 3 * C_DIM, red, tid);

  for (int c = tid; c < C_DIM; c += 256)
    out[(size_t)r * C_DIM + c] = (len > 0) ? xs[c] : 0.f;
}

extern "C" void kernel_launch(void* const* d_in, const int* in_sizes, int n_in,
                              void* d_out, int out_size, void* d_ws, size_t ws_size,
                              hipStream_t stream) {
  (void)n_in; (void)out_size; (void)ws_size;
  const float* gene   = (const float*)d_in[0];
  const int*   gidx   = (const int*)d_in[1];
  const int*   glen   = (const int*)d_in[2];
  const float* Wq     = (const float*)d_in[3];
  const float* Wk     = (const float*)d_in[4];
  const float* Wv     = (const float*)d_in[5];
  const float* Wo     = (const float*)d_in[6];
  const float* ffW    = (const float*)d_in[7];
  const float* bq     = (const float*)d_in[8];
  const float* bk     = (const float*)d_in[9];
  const float* bv     = (const float*)d_in[10];
  const float* bo     = (const float*)d_in[11];
  const float* ffb    = (const float*)d_in[12];
  const float* g1     = (const float*)d_in[13];
  const float* b1     = (const float*)d_in[14];
  const float* g2     = (const float*)d_in[15];
  const float* b2     = (const float*)d_in[16];
  const float* pmaW   = (const float*)d_in[17];
  const float* pmab   = (const float*)d_in[18];
  const float* seed   = (const float*)d_in[19];
  float* out = (float*)d_out;

  const int R = in_sizes[2];         // 4096 reactions
  const int NB = 4;
  const int nBlkW = NB * CC;         // per weight tensor

  // f16 weight staging in workspace
  half_t* w16   = (half_t*)d_ws;
  half_t* Wq16  = w16;
  half_t* Wk16  = w16 + 1 * (size_t)nBlkW;
  half_t* Wv16  = w16 + 2 * (size_t)nBlkW;
  half_t* Wo16  = w16 + 3 * (size_t)nBlkW;
  half_t* ff16  = w16 + 4 * (size_t)nBlkW;
  half_t* pma16 = w16 + 5 * (size_t)nBlkW;

  dim3 cg(1024), cb(256);
  cvt_f32_f16_kernel<<<cg, cb, 0, stream>>>(Wq,   Wq16,  nBlkW);
  cvt_f32_f16_kernel<<<cg, cb, 0, stream>>>(Wk,   Wk16,  nBlkW);
  cvt_f32_f16_kernel<<<cg, cb, 0, stream>>>(Wv,   Wv16,  nBlkW);
  cvt_f32_f16_kernel<<<cg, cb, 0, stream>>>(Wo,   Wo16,  nBlkW);
  cvt_f32_f16_kernel<<<cg, cb, 0, stream>>>(ffW,  ff16,  nBlkW);
  cvt_f32_f16_kernel<<<cg, cb, 0, stream>>>(pmaW, pma16, CC);

  (void)hipFuncSetAttribute((const void*)settransformer_kernel,
                            hipFuncAttributeMaxDynamicSharedMemorySize, SMEM_BYTES);

  settransformer_kernel<<<dim3(R), dim3(256), SMEM_BYTES, stream>>>(
      gene, gidx, glen, Wq16, Wk16, Wv16, Wo16, ff16, pma16,
      bq, bk, bv, bo, ffb, g1, b1, g2, b2, pmab, seed, out);
}
